// VAE_IGLS_8727373546072
// MI455X (gfx1250) — matrix operand order repaired and checked by reading, not compiled, not run on hardware
//
#include <hip/hip_runtime.h>
#include <hip/hip_bf16.h>

// ---------------------------------------------------------------------------
// VAE + IGLS pipeline for MI455X (gfx1250, wave32, WMMA)
// ---------------------------------------------------------------------------

#define B_OBS   128
#define N_SUBJ  16
#define Z_DIM   64
#define FLAT_N  2560
#define BN_EPS  1e-5f
#define LEAKY_S 0.01f

typedef __attribute__((ext_vector_type(16))) _Float16 v16h;
typedef __attribute__((ext_vector_type(8)))  float    v8f;

// --------------------------- direct 3D convolution --------------------------
__global__ void conv3d_k(const float* __restrict__ in, const float* __restrict__ w,
                         const float* __restrict__ bias, float* __restrict__ out,
                         int N, int Cin, int Din, int Hin, int Win,
                         int Cout, int Dout, int Hout, int Wout,
                         int stride, int pad)
{
    long long idx = (long long)blockIdx.x * blockDim.x + threadIdx.x;
    long long total = (long long)N * Cout * Dout * Hout * Wout;
    if (idx >= total) return;
    int ow = (int)(idx % Wout); long long r = idx / Wout;
    int oh = (int)(r % Hout); r /= Hout;
    int od = (int)(r % Dout); r /= Dout;
    int co = (int)(r % Cout); int n = (int)(r / Cout);
    float acc = bias[co];
    for (int ci = 0; ci < Cin; ++ci) {
        const float* ip = in + ((long long)(n * Cin + ci) * Din) * Hin * Win;
        const float* wp = w + (long long)(co * Cin + ci) * 27;
        #pragma unroll
        for (int kd = 0; kd < 3; ++kd) {
            int id = od * stride - pad + kd; if (id < 0 || id >= Din) continue;
            #pragma unroll
            for (int kh = 0; kh < 3; ++kh) {
                int ih = oh * stride - pad + kh; if (ih < 0 || ih >= Hin) continue;
                #pragma unroll
                for (int kw = 0; kw < 3; ++kw) {
                    int iw = ow * stride - pad + kw; if (iw < 0 || iw >= Win) continue;
                    acc += ip[((long long)id * Hin + ih) * Win + iw] * wp[(kd * 3 + kh) * 3 + kw];
                }
            }
        }
    }
    out[idx] = acc;
}

// --------------------- transposed 3D convolution (gather) -------------------
__global__ void convT3d_k(const float* __restrict__ in, const float* __restrict__ w,
                          const float* __restrict__ bias, float* __restrict__ out,
                          int N, int Cin, int Din, int Hin, int Win,
                          int Cout, int Dout, int Hout, int Wout,
                          int stride, int pad, int fuse_sigmoid)
{
    long long idx = (long long)blockIdx.x * blockDim.x + threadIdx.x;
    long long total = (long long)N * Cout * Dout * Hout * Wout;
    if (idx >= total) return;
    int ow = (int)(idx % Wout); long long r = idx / Wout;
    int oh = (int)(r % Hout); r /= Hout;
    int od = (int)(r % Dout); r /= Dout;
    int co = (int)(r % Cout); int n = (int)(r / Cout);
    float acc = bias[co];
    for (int ci = 0; ci < Cin; ++ci) {
        const float* ip = in + ((long long)(n * Cin + ci) * Din) * Hin * Win;
        const float* wp = w + (long long)(ci * Cout + co) * 27;   // [Cin, Cout, 3,3,3]
        #pragma unroll
        for (int kd = 0; kd < 3; ++kd) {
            int dn = od + pad - kd; if (dn < 0 || (dn % stride)) continue;
            int id = dn / stride;   if (id >= Din) continue;
            #pragma unroll
            for (int kh = 0; kh < 3; ++kh) {
                int hn = oh + pad - kh; if (hn < 0 || (hn % stride)) continue;
                int ih = hn / stride;   if (ih >= Hin) continue;
                #pragma unroll
                for (int kw = 0; kw < 3; ++kw) {
                    int wn = ow + pad - kw; if (wn < 0 || (wn % stride)) continue;
                    int iw = wn / stride;   if (iw >= Win) continue;
                    acc += ip[((long long)id * Hin + ih) * Win + iw] * wp[(kd * 3 + kh) * 3 + kw];
                }
            }
        }
    }
    if (fuse_sigmoid) acc = 1.0f / (1.0f + __expf(-acc));
    out[idx] = acc;
}

// ---------------- per-channel batch statistics (training-mode BN) -----------
__global__ void stats_k(const float* __restrict__ x, float* __restrict__ mean,
                        float* __restrict__ var, int N, int C, int spatial)
{
    int c = blockIdx.x;
    long long per = (long long)N * spatial;
    double s = 0.0, sq = 0.0;
    for (long long i = threadIdx.x; i < per; i += blockDim.x) {
        long long n = i / spatial;
        int sp = (int)(i % spatial);
        float v = x[(n * C + c) * (long long)spatial + sp];
        s += v; sq += (double)v * v;
    }
    __shared__ double ss[256], s2[256];
    ss[threadIdx.x] = s; s2[threadIdx.x] = sq;
    __syncthreads();
    for (int st = blockDim.x >> 1; st > 0; st >>= 1) {
        if ((int)threadIdx.x < st) { ss[threadIdx.x] += ss[threadIdx.x + st];
                                     s2[threadIdx.x] += s2[threadIdx.x + st]; }
        __syncthreads();
    }
    if (threadIdx.x == 0) {
        double m = ss[0] / (double)per;
        mean[c] = (float)m;
        var[c]  = (float)(s2[0] / (double)per - m * m);   // biased variance
    }
}

__global__ void bn_act_k(float* __restrict__ x, const float* __restrict__ mean,
                         const float* __restrict__ var, const float* __restrict__ g,
                         const float* __restrict__ b, int C, int spatial, long long total)
{
    long long i = (long long)blockIdx.x * blockDim.x + threadIdx.x;
    if (i >= total) return;
    int c = (int)((i / spatial) % C);
    float y = (x[i] - mean[c]) * rsqrtf(var[c] + BN_EPS) * g[c] + b[c];
    x[i] = (y >= 0.0f) ? y : LEAKY_S * y;
}

// ------------------------------ f32 -> f16 ----------------------------------
__global__ void cvt_f16_k(const float* __restrict__ in, _Float16* __restrict__ out,
                          long long n)
{
    long long i = (long long)blockIdx.x * blockDim.x + threadIdx.x;
    if (i < n) out[i] = (_Float16)in[i];
}

// -------------------- WMMA GEMM: C[M,N] = A[M,K] * W[N,K]^T + bias ----------
// One wave per 16x16 C tile; v_wmma_f32_16x16x32_f16 over K in steps of 32.
__global__ void wmma_gemm_abT_k(const _Float16* __restrict__ A,
                                const _Float16* __restrict__ W,
                                const float* __restrict__ bias,
                                float* __restrict__ C,
                                int M, int N, int K)
{
    int wavesPerBlock = blockDim.x >> 5;
    int wave = blockIdx.x * wavesPerBlock + ((int)threadIdx.x >> 5);
    int lane = (int)threadIdx.x & 31;
    int tilesN = N >> 4;
    int tiles  = (M >> 4) * tilesN;
    if (wave >= tiles) return;                 // wave-uniform: EXEC stays all-ones
    int tm = wave / tilesN, tn = wave % tilesN;

    int rowA = tm * 16 + (lane & 15);
    int kgrp = lane >> 4;                      // 0 or 1 (half-wave K group)
    int colB = tn * 16 + (lane & 15);

    v8f acc = {};
    for (int k0 = 0; k0 < K; k0 += 32) {
        const _Float16* ap = A + (long long)rowA * K + k0;
        const _Float16* bp = W + (long long)colB * K + k0 + kgrp * 16;
        __builtin_prefetch(ap + 64, 0, 1);
        __builtin_prefetch(bp + 64, 0, 1);
        v16h a, b;
        // A (16x32 f16): lanes 0-15 -> K 0..15, lanes 16-31 -> K 16..31
        #pragma unroll
        for (int v = 0; v < 8; ++v) {
            int kOff = (kgrp << 4) + (v << 1);
            a[2 * v]     = ap[kOff];
            a[2 * v + 1] = ap[kOff + 1];
        }
        // B (32x16 f16): lanes 0-15 hold K 0..15, lanes 16-31 hold K 16..31
        #pragma unroll
        for (int v = 0; v < 8; ++v) {
            b[2 * v]     = bp[2 * v];
            b[2 * v + 1] = bp[2 * v + 1];
        }
        acc = __builtin_amdgcn_wmma_f32_16x16x32_f16(false, a, false, b,
                                                     (short)0, acc, false, false);
    }
    // C tile: VGPR v -> row = tm*16 + 8*(lane/16) + v, col = tn*16 + lane%16
    int row0 = tm * 16 + (kgrp << 3);
    int col  = tn * 16 + (lane & 15);
    float bc = bias[col];
    #pragma unroll
    for (int v = 0; v < 8; ++v)
        C[(long long)(row0 + v) * N + col] = acc[v] + bc;
}

// ------------------------- IGLS: OLS init of betahat ------------------------
__global__ void beta_init_k(const float* __restrict__ z, const float* __restrict__ t,
                            float* __restrict__ beta, int Bn, int K)
{
    int k = blockIdx.x * blockDim.x + threadIdx.x;
    if (k >= K) return;
    double St = 0, Stt = 0, Sz = 0, Stz = 0;
    for (int b = 0; b < Bn; ++b) {
        double tb = t[b], zb = z[(long long)b * K + k];
        St += tb; Stt += tb * tb; Sz += zb; Stz += tb * zb;
    }
    double S1 = (double)Bn;
    double det = S1 * Stt - St * St;
    beta[2 * k]     = (float)(( Stt * Sz - St * Stz) / det);
    beta[2 * k + 1] = (float)((-St  * Sz + S1 * Stz) / det);
}

// -------------------- (zz^T zz)^-1 : 4x4 gram inverse -----------------------
__global__ void graminv_k(const int* __restrict__ sid, const float* __restrict__ t,
                          float* __restrict__ Ginv, int Bn)
{
    double g[16];
    #pragma unroll
    for (int i = 0; i < 16; ++i) g[i] = 0.0;
    for (int p = threadIdx.x; p < Bn * Bn; p += blockDim.x) {
        int b = p / Bn, c = p % Bn;
        float same = (sid[b] == sid[c]) ? 1.0f : 0.0f;
        float f[4];
        f[0] = (b == c) ? 1.0f : 0.0f;
        f[1] = same;
        f[2] = same * (t[b] + t[c]);
        f[3] = same * (t[b] * t[c]);
        #pragma unroll
        for (int i = 0; i < 4; ++i)
            #pragma unroll
            for (int j = 0; j < 4; ++j) g[i * 4 + j] += (double)f[i] * f[j];
    }
    __shared__ double sh[64][16];
    #pragma unroll
    for (int i = 0; i < 16; ++i) sh[threadIdx.x][i] = g[i];
    __syncthreads();
    if (threadIdx.x == 0) {
        double M[4][8];
        for (int i = 0; i < 4; ++i)
            for (int j = 0; j < 4; ++j) {
                double s = 0.0;
                for (int tix = 0; tix < 64; ++tix) s += sh[tix][i * 4 + j];
                M[i][j] = s;
                M[i][4 + j] = (i == j) ? 1.0 : 0.0;
            }
        for (int p = 0; p < 4; ++p) {
            double d = 1.0 / M[p][p];
            for (int j = 0; j < 8; ++j) M[p][j] *= d;
            for (int i = 0; i < 4; ++i) if (i != p) {
                double f = M[i][p];
                for (int j = 0; j < 8; ++j) M[i][j] -= f * M[p][j];
            }
        }
        for (int i = 0; i < 4; ++i)
            for (int j = 0; j < 4; ++j) Ginv[i * 4 + j] = (float)M[i][4 + j];
    }
}

// ztilde[b,k] = (beta0[k] + beta1[k]*t[b]) - z[b,k]
__global__ void ztilde_k(const float* __restrict__ z, const float* __restrict__ beta,
                         const float* __restrict__ t, float* __restrict__ zt,
                         int Bn, int K)
{
    int i = blockIdx.x * blockDim.x + threadIdx.x;
    if (i >= Bn * K) return;
    int b = i / K, k = i % K;
    zt[i] = beta[2 * k] + beta[2 * k + 1] * t[b] - z[i];
}

// sig_est = Ginv @ (zz^T ztz)  via per-subject factorized sums
__global__ void sigest_k(const float* __restrict__ zt, const float* __restrict__ t,
                         const int* __restrict__ sid, const float* __restrict__ Ginv,
                         float* __restrict__ sig, int Bn, int K, int NS)
{
    int k = blockIdx.x * blockDim.x + threadIdx.x;
    if (k >= K) return;
    double R[4] = {0, 0, 0, 0};
    for (int b = 0; b < Bn; ++b) {
        double v = zt[(long long)b * K + k];
        R[0] += v * v;
    }
    for (int s = 0; s < NS; ++s) {
        double S0 = 0, S1 = 0;
        for (int b = 0; b < Bn; ++b) {
            if (sid[b] == s) {
                double v = zt[(long long)b * K + k];
                S0 += v; S1 += (double)t[b] * v;
            }
        }
        R[1] += S0 * S0;
        R[2] += 2.0 * S0 * S1;
        R[3] += S1 * S1;
    }
    #pragma unroll
    for (int j = 0; j < 4; ++j) {
        double e = 0.0;
        #pragma unroll
        for (int i = 0; i < 4; ++i) e += (double)Ginv[j * 4 + i] * R[i];
        float ef = (float)e;
        if (ef <= 0.0f) ef = 1e-6f;               // reset_sig_est
        sig[j * K + k] = ef;                      // rows: s_e, s_a0, s_a01, s_a1
    }
}

// sigma[k,b,c] (f32) written to d_out
__global__ void sigma_k(const float* __restrict__ sig, const float* __restrict__ t,
                        const int* __restrict__ sid, float* __restrict__ out,
                        int Bn, int K)
{
    long long i = (long long)blockIdx.x * blockDim.x + threadIdx.x;
    long long total = (long long)K * Bn * Bn;
    if (i >= total) return;
    int c = (int)(i % Bn); long long r = i / Bn;
    int b = (int)(r % Bn); int k = (int)(r / Bn);
    float same = (sid[b] == sid[c]) ? 1.0f : 0.0f;
    float v = sig[0 * K + k] * ((b == c) ? 1.0f : 0.0f)
            + same * (sig[1 * K + k]
                      + sig[2 * K + k] * (t[b] + t[c])
                      + sig[3 * K + k] * (t[b] * t[c]));
    out[i] = v;
}

// f64 in-place Gauss-Jordan inversion of each 128x128 sigma, in 128 KB LDS.
// One workgroup (128 threads, thread = row) per k. CDNA5: 320 KB LDS/WGP.
__global__ void siginv_k(const float* __restrict__ sigma, float* __restrict__ inv, int Bn)
{
    __shared__ double As[128 * 128];
    int k = blockIdx.x;
    const float* S = sigma + (long long)k * Bn * Bn;
    for (int i = threadIdx.x; i < Bn * Bn; i += blockDim.x) As[i] = (double)S[i];
    __syncthreads();
    int tid = (int)threadIdx.x;
    for (int p = 0; p < Bn; ++p) {
        double piv = As[p * Bn + p];
        double d = 1.0 / piv;
        __syncthreads();
        if (tid == p) As[p * Bn + p] = d;
        else          As[p * Bn + tid] *= d;
        __syncthreads();
        if (tid != p) {
            double f = As[tid * Bn + p];
            for (int j = 0; j < Bn; ++j)
                if (j != p) As[tid * Bn + j] -= f * As[p * Bn + j];
            As[tid * Bn + p] = -f * d;
        }
        __syncthreads();
    }
    for (int i = threadIdx.x; i < Bn * Bn; i += blockDim.x)
        inv[(long long)k * Bn * Bn + i] = (float)As[i];
}

// betahat = (X^T S^-1 X)^-1 (X^T S^-1 z) per k
__global__ void beta_final_k(const float* __restrict__ inv, const float* __restrict__ z,
                             const float* __restrict__ t, float* __restrict__ betaOut,
                             float* __restrict__ betaWs, int Bn, int K)
{
    int k = blockIdx.x;
    const float* S = inv + (long long)k * Bn * Bn;
    double a00 = 0, a01 = 0, a10 = 0, a11 = 0, r0 = 0, r1 = 0;
    for (int p = threadIdx.x; p < Bn * Bn; p += blockDim.x) {
        int b = p / Bn, c = p % Bn;
        double s = S[p], tb = t[b], tc = t[c], zc = z[(long long)c * K + k];
        a00 += s;           a01 += s * tc;
        a10 += s * tb;      a11 += s * tb * tc;
        r0  += s * zc;      r1  += s * tb * zc;
    }
    __shared__ double sh[6][256];
    sh[0][threadIdx.x] = a00; sh[1][threadIdx.x] = a01; sh[2][threadIdx.x] = a10;
    sh[3][threadIdx.x] = a11; sh[4][threadIdx.x] = r0;  sh[5][threadIdx.x] = r1;
    __syncthreads();
    for (int st = blockDim.x >> 1; st > 0; st >>= 1) {
        if ((int)threadIdx.x < st)
            #pragma unroll
            for (int q = 0; q < 6; ++q) sh[q][threadIdx.x] += sh[q][threadIdx.x + st];
        __syncthreads();
    }
    if (threadIdx.x == 0) {
        double A00 = sh[0][0], A01 = sh[1][0], A10 = sh[2][0], A11 = sh[3][0];
        double R0 = sh[4][0], R1 = sh[5][0];
        double det = A00 * A11 - A01 * A10;
        float b0 = (float)(( A11 * R0 - A01 * R1) / det);
        float b1 = (float)((-A10 * R0 + A00 * R1) / det);
        betaOut[2 * k] = b0; betaOut[2 * k + 1] = b1;
        betaWs[2 * k]  = b0; betaWs[2 * k + 1]  = b1;
    }
}

// mu, z_hat (reparameterization)
__global__ void mu_zhat_k(const float* __restrict__ beta, const float* __restrict__ sig,
                          const float* __restrict__ t, const int* __restrict__ sid,
                          const float* __restrict__ e0, const float* __restrict__ e1,
                          const float* __restrict__ ee,
                          float* __restrict__ muOut, float* __restrict__ zhOut,
                          float* __restrict__ zhWs, int Bn, int K)
{
    int i = blockIdx.x * blockDim.x + threadIdx.x;
    if (i >= Bn * K) return;
    int b = i / K, k = i % K;
    float m = beta[2 * k] + beta[2 * k + 1] * t[b];
    muOut[(long long)k * Bn + b] = m;
    int s = sid[b];
    float a0 = e0[(long long)s * K + k] * sqrtf(sig[1 * K + k]);
    float a1 = e1[(long long)s * K + k] * sqrtf(sig[3 * K + k]) * t[b];
    float e  = ee[i] * sqrtf(sig[0 * K + k]);
    float zh = m + a0 + a1 + e;
    zhOut[i] = zh; zhWs[i] = zh;
}

__global__ void igls_vars_k(const float* __restrict__ sig, float* __restrict__ out, int K)
{
    int k = blockIdx.x * blockDim.x + threadIdx.x;
    if (k >= K) return;
    out[0 * K + k] = sig[1 * K + k];   // sig_rand[0,0] = s_a0
    out[1 * K + k] = sig[3 * K + k];   // sig_rand[1,1] = s_a1
    out[2 * K + k] = sig[0 * K + k];   // s_e
}

// ---------------------------------------------------------------------------
extern "C" void kernel_launch(void* const* d_in, const int* in_sizes, int n_in,
                              void* d_out, int out_size, void* d_ws, size_t ws_size,
                              hipStream_t stream)
{
    (void)in_sizes; (void)n_in; (void)out_size; (void)ws_size;

    const float* x   = (const float*)d_in[0];
    const int*   sid = (const int*)  d_in[1];
    const float* t   = (const float*)d_in[2];
    const float* e0  = (const float*)d_in[3];
    const float* e1  = (const float*)d_in[4];
    const float* ee  = (const float*)d_in[5];
    // params: encoder layer i -> w:6+4i b:7+4i g:8+4i bb:9+4i
    const float* zw  = (const float*)d_in[30];
    const float* zb  = (const float*)d_in[31];
    const float* dlw = (const float*)d_in[32];
    const float* dlb = (const float*)d_in[33];
    // decoder layer j (1..4) -> dw:34+4(j-1) db:35+4(j-1) dg:36+4(j-1) dbb:37+4(j-1)

    // ---- output offsets (floats) ----
    float* outF = (float*)d_out;
    const long long OFF_D     = 0;                       // [128,1,56,48,48]
    const long long OFF_Z     = 16515072;                // [128,64]
    const long long OFF_ZHAT  = OFF_Z + 8192;            // [128,64]
    const long long OFF_SIGMA = OFF_ZHAT + 8192;         // [64,128,128]
    const long long OFF_MU    = OFF_SIGMA + 1048576;     // [64,128]
    const long long OFF_BETA  = OFF_MU + 8192;           // [64,2,1]
    const long long OFF_IGLS  = OFF_BETA + 128;          // [3,64]

    // ---- workspace bump allocator ----
    char* base = (char*)d_ws;
    size_t off = 0;
    auto alloc = [&](size_t bytes) -> void* {
        void* p = base + off;
        off += bytes;
        off = (off + 255) & ~(size_t)255;
        return p;
    };
    float*    s1    = (float*)   alloc((size_t)5242880 * 4);
    float*    s2    = (float*)   alloc((size_t)16515072 * 4);
    float*    meanB = (float*)   alloc(64 * 4);
    float*    varB  = (float*)   alloc(64 * 4);
    _Float16* h16   = (_Float16*)alloc((size_t)327680 * 2);
    _Float16* wz16  = (_Float16*)alloc((size_t)163840 * 2);
    float*    betaW = (float*)   alloc(128 * 4);
    float*    GinvW = (float*)   alloc(16 * 4);
    float*    ztW   = (float*)   alloc(8192 * 4);
    float*    sigW  = (float*)   alloc(256 * 4);
    float*    invW  = (float*)   alloc((size_t)4194304 * 4);
    float*    zhW   = (float*)   alloc(8192 * 4);
    _Float16* zh16  = (_Float16*)alloc(8192 * 2);
    _Float16* wd16  = (_Float16*)alloc((size_t)163840 * 2);
    float*    d0W   = (float*)   alloc((size_t)327680 * 4);
    float*    d1W   = (float*)   alloc((size_t)516096 * 4);
    float*    d2W   = (float*)   alloc((size_t)4128768 * 4);

    auto nb = [](long long n, int b) { return (unsigned)((n + b - 1) / b); };

    // ---------------- Encoder: 6x (conv, BN(batch stats), lrelu) ------------
    struct ECfg { int ci, co, di, hi, wi, dd, hh, ww, st; };
    const ECfg enc[6] = {
        { 1,  8, 40, 32, 32, 20, 16, 16, 2},
        { 8,  8, 20, 16, 16, 20, 16, 16, 1},
        { 8, 16, 20, 16, 16, 10,  8,  8, 2},
        {16, 16, 10,  8,  8, 10,  8,  8, 1},
        {16, 32, 10,  8,  8,  5,  4,  4, 2},
        {32, 32,  5,  4,  4,  5,  4,  4, 1},
    };
    for (int i = 0; i < 6; ++i) {
        const ECfg& L = enc[i];
        const float* w  = (const float*)d_in[6 + 4 * i];
        const float* bb = (const float*)d_in[7 + 4 * i];
        const float* g  = (const float*)d_in[8 + 4 * i];
        const float* bn = (const float*)d_in[9 + 4 * i];
        const float* src = (i == 0) ? x : ((i & 1) ? s1 : s2);
        float* dst = (i & 1) ? s2 : s1;
        int spatial = L.dd * L.hh * L.ww;
        long long total = (long long)B_OBS * L.co * spatial;
        conv3d_k<<<nb(total, 256), 256, 0, stream>>>(src, w, bb, dst,
            B_OBS, L.ci, L.di, L.hi, L.wi, L.co, L.dd, L.hh, L.ww, L.st, 1);
        stats_k<<<L.co, 256, 0, stream>>>(dst, meanB, varB, B_OBS, L.co, spatial);
        bn_act_k<<<nb(total, 256), 256, 0, stream>>>(dst, meanB, varB, g, bn,
                                                     L.co, spatial, total);
    }
    // h = s2 flattened to [128, 2560]

    // ---------------- z_ijk = h @ zw^T + zb (WMMA) ---------------------------
    cvt_f16_k<<<nb(327680, 256), 256, 0, stream>>>(s2, h16, 327680);
    cvt_f16_k<<<nb(163840, 256), 256, 0, stream>>>(zw, wz16, 163840);
    {
        int tiles = (B_OBS / 16) * (Z_DIM / 16);      // 32
        wmma_gemm_abT_k<<<nb(tiles, 8), 256, 0, stream>>>(h16, wz16, zb,
            outF + OFF_Z, B_OBS, Z_DIM, FLAT_N);
    }
    const float* zout = outF + OFF_Z;

    // ---------------- IGLS -------------------------------------------------
    beta_init_k<<<1, 64, 0, stream>>>(zout, t, betaW, B_OBS, Z_DIM);
    graminv_k<<<1, 64, 0, stream>>>(sid, t, GinvW, B_OBS);
    ztilde_k<<<nb(B_OBS * Z_DIM, 256), 256, 0, stream>>>(zout, betaW, t, ztW,
                                                         B_OBS, Z_DIM);
    sigest_k<<<1, 64, 0, stream>>>(ztW, t, sid, GinvW, sigW, B_OBS, Z_DIM, N_SUBJ);
    sigma_k<<<nb((long long)Z_DIM * B_OBS * B_OBS, 256), 256, 0, stream>>>(
        sigW, t, sid, outF + OFF_SIGMA, B_OBS, Z_DIM);
    siginv_k<<<Z_DIM, 128, 0, stream>>>(outF + OFF_SIGMA, invW, B_OBS);
    beta_final_k<<<Z_DIM, 256, 0, stream>>>(invW, zout, t, outF + OFF_BETA, betaW,
                                            B_OBS, Z_DIM);
    mu_zhat_k<<<nb(B_OBS * Z_DIM, 256), 256, 0, stream>>>(betaW, sigW, t, sid,
        e0, e1, ee, outF + OFF_MU, outF + OFF_ZHAT, zhW, B_OBS, Z_DIM);
    igls_vars_k<<<1, 64, 0, stream>>>(sigW, outF + OFF_IGLS, Z_DIM);

    // ---------------- Decoder ----------------------------------------------
    // d0 = z_hat @ dlw^T + dlb  (WMMA)
    cvt_f16_k<<<nb(8192, 256), 256, 0, stream>>>(zhW, zh16, 8192);
    cvt_f16_k<<<nb(163840, 256), 256, 0, stream>>>(dlw, wd16, 163840);
    {
        int tiles = (B_OBS / 16) * (FLAT_N / 16);     // 1280
        wmma_gemm_abT_k<<<nb(tiles, 8), 256, 0, stream>>>(zh16, wd16, dlb,
            d0W, B_OBS, FLAT_N, Z_DIM);
    }

    struct DCfg { int ci, co, di, hi, wi, dd, hh, ww, st, pd; };
    const DCfg dec[4] = {
        {32, 16,  5,  4,  4,  7,  6,  6, 1, 0},
        {16, 16,  7,  6,  6, 14, 12, 12, 2, 1},
        {16,  8, 14, 12, 12, 28, 24, 24, 2, 1},
        { 8,  1, 28, 24, 24, 56, 48, 48, 2, 1},
    };
    float* dbufs_in[4]  = { d0W, d1W, d2W, s2 };
    float* dbufs_out[4] = { d1W, d2W, s2, outF + OFF_D };
    for (int j = 0; j < 4; ++j) {
        const DCfg& L = dec[j];
        const float* w  = (const float*)d_in[34 + 4 * j];
        const float* bb = (const float*)d_in[35 + 4 * j];
        int spatial = L.dd * L.hh * L.ww;
        long long total = (long long)B_OBS * L.co * spatial;
        convT3d_k<<<nb(total, 256), 256, 0, stream>>>(dbufs_in[j], w, bb, dbufs_out[j],
            B_OBS, L.ci, L.di, L.hi, L.wi, L.co, L.dd, L.hh, L.ww, L.st, L.pd,
            (j == 3) ? 1 : 0);
        if (j < 3) {
            const float* g  = (const float*)d_in[36 + 4 * j];
            const float* bn = (const float*)d_in[37 + 4 * j];
            stats_k<<<L.co, 256, 0, stream>>>(dbufs_out[j], meanB, varB,
                                              B_OBS, L.co, spatial);
            bn_act_k<<<nb(total, 256), 256, 0, stream>>>(dbufs_out[j], meanB, varB,
                g, bn, L.co, spatial, total);
        }
    }
}
